// MoELogisticRegression_11029476016647
// MI455X (gfx1250) — matrix operand order repaired and checked by reading, not compiled, not run on hardware
//
#include <hip/hip_runtime.h>
#include <stdint.h>

// ---------- problem constants ----------
#define B_ROWS 16384
#define DDIM   4096
#define EEXP   64
#define TOPK   8

// ---------- tiling ----------
#define WAVES_PER_WG   4
#define THREADS        (WAVES_PER_WG * 32)          // 128
#define ROWS_PER_WAVE  16
#define ROWS_PER_WG    (WAVES_PER_WG * ROWS_PER_WAVE)  // 64 -> 256 workgroups
#define KC             64                            // K-chunk staged in LDS
#define NCHUNK         (DDIM / KC)                   // 64
#define ROW_PITCH_DW   (KC + 4)                      // 68 dwords (TDM pad: +4 DW/row)
#define ROW_PITCH_B    (ROW_PITCH_DW * 4)            // 272 B  (bank stride 4 -> conflict free)
#define MTX_BYTES      (64 * ROW_PITCH_B)            // 17408 B per weight matrix tile
#define BUF_BYTES      (128 * ROW_PITCH_B)           // 34816 B per chunk buffer
#define SMEM_BYTES     (2 * BUF_BYTES)               // double-buffered: 69632 B
#define EPI_PITCH_DW   132                           // epilogue logits pitch (132%64==4)

typedef __attribute__((ext_vector_type(4)))  float    f32x4;
typedef __attribute__((ext_vector_type(8)))  float    f32x8;
typedef __attribute__((ext_vector_type(8)))  float    v8f;
typedef __attribute__((ext_vector_type(16))) __bf16   v16bf;
typedef __attribute__((ext_vector_type(4)))  uint32_t u32x4;
typedef __attribute__((ext_vector_type(8)))  uint32_t u32x8;

// Issue one TDM 2D tile load: rows=64, tile_dim0=KC fp32 elements, row stride DDIM,
// LDS rows padded to ROW_PITCH_DW dwords via pad_enable.
__device__ __forceinline__ void tdm_load_tile(uint32_t lds_byte_off, const float* gsrc)
{
    uint64_t ga = (uint64_t)(uintptr_t)gsrc;
    u32x4 g0;
    g0[0] = 1u;                                         // count=1 (valid), user mode
    g0[1] = lds_byte_off;                               // lds_addr
    g0[2] = (uint32_t)ga;                               // global_addr[31:0]
    g0[3] = ((uint32_t)(ga >> 32) & 0x01FFFFFFu)        // global_addr[56:32]
          | (2u << 30);                                 // type = 2 ("image")
    u32x8 g1;
    g1[0] = (2u << 16)                                  // data_size = 4 bytes
          | (1u << 20)                                  // pad_enable
          | (5u << 22)                                  // pad_interval: 64 DWORDs
          | (3u << 25);                                 // pad_amount:   4 DWORDs
    g1[1] = ((uint32_t)DDIM & 0xFFFFu) << 16;           // tensor_dim0[15:0]
    g1[2] = ((uint32_t)DDIM >> 16) | (64u << 16);       // tensor_dim0[31:16] | tensor_dim1[15:0]
    g1[3] = ((uint32_t)KC) << 16;                       // tensor_dim1[31:16]=0 | tile_dim0=KC
    g1[4] = 64u;                                        // tile_dim1=64 rows, tile_dim2=0
    g1[5] = (uint32_t)DDIM;                             // tensor_dim0_stride[31:0]
    g1[6] = 0u;
    g1[7] = 0u;
    asm volatile("tensor_load_to_lds %0, %1" :: "s"(g0), "s"(g1) : "memory");
}

__device__ __forceinline__ v16bf cvt16(f32x8 lo, f32x8 hi)
{
    v16bf r;
#pragma unroll
    for (int i = 0; i < 8; ++i) { r[i] = (__bf16)lo[i]; r[8 + i] = (__bf16)hi[i]; }
    return r;
}

__global__ void __launch_bounds__(THREADS)
moe_fused_kernel(const float* __restrict__ x,
                 const float* __restrict__ noise_w,
                 const float* __restrict__ noise_b,
                 const float* __restrict__ expert_w,
                 const float* __restrict__ expert_b,
                 float* __restrict__ out)
{
    extern __shared__ char smem[];
    const int tid  = threadIdx.x;
    const int wave = tid >> 5;
    const int lane = tid & 31;
    const int l15  = lane & 15;
    const int sel  = lane >> 4;          // K-group select for WMMA fragments

    const int rowBaseWG = blockIdx.x * ROWS_PER_WG;
    const int row       = rowBaseWG + wave * ROWS_PER_WAVE + l15;
    const float* xr     = x + (size_t)row * DDIM;

    // low 32 bits of a shared-memory flat address == LDS byte offset (dynamic LDS starts at 0)
    const uint32_t smem_base = (uint32_t)(uintptr_t)smem;

    v8f acc[8];
#pragma unroll
    for (int t = 0; t < 8; ++t)
#pragma unroll
        for (int r = 0; r < 8; ++r) acc[t][r] = 0.0f;

    // ---- prologue: DMA chunk 0 (noise_w rows -> buf0[0:64), expert_w -> buf0[64:128)) ----
    if (wave == 0) {
        tdm_load_tile(smem_base + 0,         noise_w);
        tdm_load_tile(smem_base + MTX_BYTES, expert_w);
    }

#pragma unroll 1
    for (int c = 0; c < NCHUNK; ++c) {
        __builtin_amdgcn_s_wait_tensorcnt(0);   // chunk c landed (no-op for non-issuing waves)
        __syncthreads();                        // visible to all; prev buffer fully consumed
        if (wave == 0 && (c + 1) < NCHUNK) {    // prefetch chunk c+1 into the other buffer
            uint32_t boff = smem_base + (uint32_t)(((c + 1) & 1) * BUF_BYTES);
            tdm_load_tile(boff,             noise_w  + (size_t)(c + 1) * KC);
            tdm_load_tile(boff + MTX_BYTES, expert_w + (size_t)(c + 1) * KC);
        }

        const char* buf = smem + (size_t)((c & 1) * BUF_BYTES);
        const int   kb  = c * KC;

#pragma unroll
        for (int s = 0; s < 2; ++s) {           // two 16x16x32 K-steps per chunk
            // A fragment (x): lane<16 holds K {0..15}, lane>=16 holds {16..31}
            f32x8 alo = *(const f32x8*)(xr + kb + 32 * s + sel * 16);
            f32x8 ahi = *(const f32x8*)(xr + kb + 32 * s + 8 + sel * 16);
            v16bf av  = cvt16(alo, ahi);

#pragma unroll
            for (int t = 0; t < 8; ++t) {       // 4 noise tiles + 4 expert tiles (N=128 total)
                const int   n64  = 16 * (t & 3) + l15;
                const char* rowp = buf + (t >> 2) * MTX_BYTES + n64 * ROW_PITCH_B;
                const f32x4* rp  = (const f32x4*)(rowp + (size_t)(32 * s + 16 * sel) * 4);
                f32x4 b0 = rp[0], b1 = rp[1], b2 = rp[2], b3 = rp[3];
                v16bf bv;
#pragma unroll
                for (int i = 0; i < 4; ++i) {
                    bv[i]      = (__bf16)b0[i];
                    bv[4 + i]  = (__bf16)b1[i];
                    bv[8 + i]  = (__bf16)b2[i];
                    bv[12 + i] = (__bf16)b3[i];
                }
                acc[t] = __builtin_amdgcn_wmma_f32_16x16x32_bf16(
                             false, av, false, bv, (short)0, acc[t], false, false);
            }
        }
    }

    // ---- bias add (per output column) ----
#pragma unroll
    for (int t = 0; t < 8; ++t) {
        const int n64 = 16 * (t & 3) + l15;
        float bias = (t < 4) ? noise_b[n64] : expert_b[n64];
#pragma unroll
        for (int r = 0; r < 8; ++r) acc[t][r] += bias;
    }

    // ---- park logits in LDS: [64 rows][132 dw] : cols 0..63 noise, 64..127 expert ----
    __syncthreads();                            // weight buffers no longer needed
    float* epi = (float*)smem;
#pragma unroll
    for (int t = 0; t < 8; ++t) {
        const int col = 16 * (t & 3) + l15 + 64 * (t >> 2);
#pragma unroll
        for (int r = 0; r < 8; ++r) {
            const int m = (lane < 16) ? r : (r + 8);
            epi[(wave * ROWS_PER_WAVE + m) * EPI_PITCH_DW + col] = acc[t][r];
        }
    }
    __syncthreads();

    // ---- per-row epilogue: top-8 of noise logits -> softmax gates -> dot with sigmoid(expert) ----
    if (tid < ROWS_PER_WG) {
        const float* rp = epi + tid * EPI_PITCH_DW;
        float v[EEXP];                           // register-resident, static indexing only
#pragma unroll
        for (int j = 0; j < EEXP; ++j) v[j] = rp[j];

        unsigned long long mask = 0ull;
        float topmax = -3.4e38f;
#pragma unroll 1
        for (int p = 0; p < TOPK; ++p) {
            float mv = -3.4e38f; int mj = 0;
#pragma unroll
            for (int j = 0; j < EEXP; ++j) {
                bool take = (((mask >> j) & 1ull) == 0ull) && (v[j] > mv);
                mv = take ? v[j] : mv;
                mj = take ? j    : mj;
            }
            mask |= (1ull << mj);
            topmax = (p == 0) ? mv : topmax;
        }

        float denom = 0.0f, num = 0.0f;
#pragma unroll
        for (int j = 0; j < EEXP; ++j) {
            if ((mask >> j) & 1ull) {
                float g   = __expf(v[j] - topmax);
                float z   = rp[EEXP + j];
                float sig = 1.0f / (1.0f + __expf(-z));
                denom += g;
                num   += g * sig;
            }
        }
        out[rowBaseWG + tid] = num / denom;
    }
}

extern "C" void kernel_launch(void* const* d_in, const int* in_sizes, int n_in,
                              void* d_out, int out_size, void* d_ws, size_t ws_size,
                              hipStream_t stream)
{
    (void)in_sizes; (void)n_in; (void)out_size; (void)d_ws; (void)ws_size;
    const float* x        = (const float*)d_in[0];
    // d_in[1] (noise), d_in[2] (router_w), d_in[3] (router_b) are provably dead:
    // noisy_logits is never consumed by the reference's output.
    const float* noise_w  = (const float*)d_in[4];
    const float* noise_b  = (const float*)d_in[5];
    const float* expert_w = (const float*)d_in[6];
    const float* expert_b = (const float*)d_in[7];
    float* out = (float*)d_out;

    dim3 grid(B_ROWS / ROWS_PER_WG);   // 256 workgroups
    dim3 block(THREADS);               // 128 threads = 4 waves
    hipLaunchKernelGGL(moe_fused_kernel, grid, block, SMEM_BYTES, stream,
                       x, noise_w, noise_b, expert_w, expert_b, out);
}
